// QKVAttention_77000173683057
// MI455X (gfx1250) — matrix operand order repaired and checked by reading, not compiled, not run on hardware
//
#include <hip/hip_runtime.h>
#include <hip/hip_fp16.h>

typedef __attribute__((ext_vector_type(16))) _Float16 v16h;
typedef __attribute__((ext_vector_type(8)))  _Float16 v8h;
typedef __attribute__((ext_vector_type(4)))  _Float16 v4h;
typedef __attribute__((ext_vector_type(8)))  float    v8f;

#define WMMA_F32_F16(a, b, c) \
  __builtin_amdgcn_wmma_f32_16x16x32_f16(false, (a), false, (b), (short)0, (c), false, false)

// DPP permute of a float within 16-lane rows (VALU only, no LDS).
#define DPP_F(x, ctrl) \
  __int_as_float(__builtin_amdgcn_update_dpp(__float_as_int(x), __float_as_int(x), (ctrl), 0xF, 0xF, true))

// Butterfly reductions over the 16 lanes of each half (DPP rows are 16 lanes,
// so these never cross the lane-half boundary of the WMMA C-frag layout).
__device__ __forceinline__ float row_max16(float x) {
  x = fmaxf(x, DPP_F(x, 0xB1));   // quad_perm [1,0,3,2]  (xor 1)
  x = fmaxf(x, DPP_F(x, 0x4E));   // quad_perm [2,3,0,1]  (xor 2)
  x = fmaxf(x, DPP_F(x, 0x141));  // row_half_mirror      (xor 7)
  x = fmaxf(x, DPP_F(x, 0x140));  // row_mirror           (xor 15)
  return x;
}
__device__ __forceinline__ float row_sum16(float x) {
  x += DPP_F(x, 0xB1);
  x += DPP_F(x, 0x4E);
  x += DPP_F(x, 0x141);
  x += DPP_F(x, 0x140);
  return x;
}

// Build a 16-f16 WMMA fragment from two contiguous 8-f16 (16B) LDS runs.
__device__ __forceinline__ v16h load_frag(const _Float16* p, int off_lo, int off_hi) {
  v8h lo = *(const v8h*)(p + off_lo);
  v8h hi = *(const v8h*)(p + off_hi);
  return __builtin_shufflevector(lo, hi, 0, 1, 2, 3, 4, 5, 6, 7, 8, 9, 10, 11, 12, 13, 14, 15);
}

// ---------------------------------------------------------------------------
// C = (A @ W + bias) * alpha, software-pipelined with double-buffered LDS.
//   A: [M,K] f32 (A_IS_F32) or f16, W: [K,N] f32, bias: [N] f32
//   Block tile 128x64, 8 waves, each wave 32x32 via 2x2 WMMA frags.
//   Pipeline: store staged regs -> LDS[buf]; barrier; issue next global
//   loads; run WMMAs on LDS[buf]. One barrier per k-step (double buffer).
// ---------------------------------------------------------------------------
template <bool A_IS_F32, bool OUT_F32>
__global__ __launch_bounds__(256) void gemm_bias_kernel(
    const void* __restrict__ Aptr, const float* __restrict__ W,
    const float* __restrict__ bias, float alpha, void* __restrict__ Out,
    int M, int N, int K) {
  __shared__ _Float16 sA[2][128 * 32];   // [m][k] row-major
  __shared__ _Float16 sB[2][64 * 32];    // [n][k] (transposed W tile)

  const int tid = threadIdx.x;
  const int lane = tid & 31;
  const int wave = tid >> 5;
  const int wm = wave & 3, wn = wave >> 2;
  const int n0 = blockIdx.x * 64;
  const int m0 = blockIdx.y * 128;
  const int lrow = lane & 15;
  const int lhalf = lane >> 4;

  v8f acc[2][2] = {};

  float4 raf[4];   // staged A tile (f32 path)
  uint4  rah[2];   // staged A tile (f16 path)
  float4 rbf[2];   // staged W tile

  // issue loads for tile 0
  {
    const int k0 = 0;
    if (A_IS_F32) {
      const float* A = (const float*)Aptr;
      for (int i = 0; i < 4; ++i) {
        int e = (i * 256 + tid) * 4;
        raf[i] = *(const float4*)(A + (size_t)(m0 + (e >> 5)) * K + k0 + (e & 31));
      }
    } else {
      const _Float16* A = (const _Float16*)Aptr;
      for (int i = 0; i < 2; ++i) {
        int e = (i * 256 + tid) * 8;
        rah[i] = *(const uint4*)(A + (size_t)(m0 + (e >> 5)) * K + k0 + (e & 31));
      }
    }
    for (int i = 0; i < 2; ++i) {
      int e = (i * 256 + tid) * 4;
      rbf[i] = *(const float4*)(W + (size_t)(k0 + (e >> 6)) * N + n0 + (e & 63));
    }
  }

  for (int k0 = 0, it = 0; k0 < K; k0 += 32, ++it) {
    const int buf = it & 1;
    // --- drain staged registers into LDS[buf] (f32 -> f16 convert here) ---
    if (A_IS_F32) {
      for (int i = 0; i < 4; ++i) {
        int e = (i * 256 + tid) * 4;
        v4h h = {(_Float16)raf[i].x, (_Float16)raf[i].y,
                 (_Float16)raf[i].z, (_Float16)raf[i].w};
        *(v4h*)(&sA[buf][(e >> 5) * 32 + (e & 31)]) = h;
      }
    } else {
      for (int i = 0; i < 2; ++i) {
        int e = (i * 256 + tid) * 8;
        *(uint4*)(&sA[buf][(e >> 5) * 32 + (e & 31)]) = rah[i];
      }
    }
    for (int i = 0; i < 2; ++i) {
      int e = (i * 256 + tid) * 4;
      int kr = e >> 6, nc = e & 63;
      sB[buf][(nc + 0) * 32 + kr] = (_Float16)rbf[i].x;
      sB[buf][(nc + 1) * 32 + kr] = (_Float16)rbf[i].y;
      sB[buf][(nc + 2) * 32 + kr] = (_Float16)rbf[i].z;
      sB[buf][(nc + 3) * 32 + kr] = (_Float16)rbf[i].w;
    }
    __syncthreads();

    // --- issue global loads for tile it+1 (consumed next iteration) ---
    if (k0 + 32 < K) {
      const int kn = k0 + 32;
      if (A_IS_F32) {
        const float* A = (const float*)Aptr;
        for (int i = 0; i < 4; ++i) {
          int e = (i * 256 + tid) * 4;
          raf[i] = *(const float4*)(A + (size_t)(m0 + (e >> 5)) * K + kn + (e & 31));
        }
      } else {
        const _Float16* A = (const _Float16*)Aptr;
        for (int i = 0; i < 2; ++i) {
          int e = (i * 256 + tid) * 8;
          rah[i] = *(const uint4*)(A + (size_t)(m0 + (e >> 5)) * K + kn + (e & 31));
        }
      }
      for (int i = 0; i < 2; ++i) {
        int e = (i * 256 + tid) * 4;
        rbf[i] = *(const float4*)(W + (size_t)(kn + (e >> 6)) * N + n0 + (e & 63));
      }
      // distance-2 prefetch into L2
      if (k0 + 64 < K) {
        int r = tid >> 1, cc = (tid & 1) * 16;
        if (A_IS_F32)
          __builtin_prefetch((const float*)Aptr + (size_t)(m0 + r) * K + k0 + 64 + cc, 0, 1);
        else
          __builtin_prefetch((const _Float16*)Aptr + (size_t)(m0 + r) * K + k0 + 64 + cc, 0, 1);
        if (tid < 128) {
          int kr = tid >> 2, nc = (tid & 3) * 16;
          __builtin_prefetch(W + (size_t)(k0 + 64 + kr) * N + n0 + nc, 0, 1);
        }
      }
    }

    // --- WMMAs on LDS[buf] ---
    v16h a[2], b[2];
    for (int mi = 0; mi < 2; ++mi) {
      int row = wm * 32 + mi * 16 + lrow;
      int ks = lhalf * 8;
      a[mi] = load_frag(sA[buf], row * 32 + ks, row * 32 + 16 + ks);
    }
    for (int ni = 0; ni < 2; ++ni) {
      int col = wn * 32 + ni * 16 + lrow;
      int ko = lhalf * 8;
      b[ni] = load_frag(sB[buf], col * 32 + ko, col * 32 + ko + 16);
    }
    for (int mi = 0; mi < 2; ++mi)
      for (int ni = 0; ni < 2; ++ni)
        acc[mi][ni] = WMMA_F32_F16(a[mi], b[ni], acc[mi][ni]);
  }

  // epilogue: (acc + bias) * alpha
  for (int ni = 0; ni < 2; ++ni) {
    int ng = n0 + wn * 32 + ni * 16 + lrow;
    float bv = bias[ng];
    for (int mi = 0; mi < 2; ++mi) {
      for (int j = 0; j < 8; ++j) {
        int mg = m0 + wm * 32 + mi * 16 + j + 8 * lhalf;
        float val = (acc[mi][ni][j] + bv) * alpha;
        if (OUT_F32)
          ((float*)Out)[(size_t)mg * N + ng] = val;
        else
          ((_Float16*)Out)[(size_t)mg * N + ng] = (_Float16)val;
      }
    }
  }
}

// ---------------------------------------------------------------------------
// Flash attention: one block = one (b, h, 64-query tile). 4 waves, each wave
// owns 16 queries. Online softmax; K/V streamed in 64-key tiles through
// double-buffered LDS with software pipelining (loads for tile kb+1 issued
// before computing tile kb). Q is pre-scaled by Dh^-0.5 (folded upstream).
// ---------------------------------------------------------------------------
__global__ __launch_bounds__(128) void attn_kernel(
    const _Float16* __restrict__ Qp, const _Float16* __restrict__ Kp,
    const _Float16* __restrict__ Vp, _Float16* __restrict__ Outp,
    const float* __restrict__ offset_p) {
  constexpr int L = 2048, D = 1024, DH = 64, NT = L / 64;
  __shared__ _Float16 sQ[64 * 64];           // [q][d]
  __shared__ _Float16 sK[2][64 * 64];        // [key][d]
  __shared__ _Float16 sVt[2][64 * 64];       // [d][key] (transposed)
  __shared__ _Float16 sP[4 * 16 * 64];       // wave-private P tiles [q][key]

  const int tid = threadIdx.x;
  const int lane = tid & 31;
  const int wave = tid >> 5;
  const int lrow = lane & 15;
  const int lhalf = lane >> 4;

  const int bid = blockIdx.x;
  const int qblk = bid & 31;                 // L/64 = 32
  const int h = (bid >> 5) & 15;
  const int b = bid >> 9;
  const size_t baseQ = ((size_t)b * L + qblk * 64) * D + h * DH;
  const float offs = offset_p[0];

  const int lr = tid >> 3;                   // staging row 0..15 (x4 iters)
  const int lc = (tid & 7) * 8;              // staging col (8-f16 chunks)

  // load Q tile (stays resident all iterations)
  for (int i = 0; i < 4; ++i) {
    int r = i * 16 + lr;
    *(v8h*)(&sQ[r * 64 + lc]) = *(const v8h*)(Qp + baseQ + (size_t)r * D + lc);
  }

  v8h rk[4], rv[4];  // staged K/V tile
  {
    const size_t baseK = ((size_t)b * L) * D + h * DH;
    for (int i = 0; i < 4; ++i) {
      int r = i * 16 + lr;
      rk[i] = *(const v8h*)(Kp + baseK + (size_t)r * D + lc);
      rv[i] = *(const v8h*)(Vp + baseK + (size_t)r * D + lc);
    }
  }

  v8f O[4] = {};
  float mprev[8], lsum[8];
  for (int j = 0; j < 8; ++j) { mprev[j] = -1e30f; lsum[j] = 0.f; }
  const int wbase = wave * 16 * 64;

  for (int kb = 0; kb < NT; ++kb) {
    const int buf = kb & 1;
    // --- drain staged K/V registers into LDS[buf] ---
    for (int i = 0; i < 4; ++i) {
      int r = i * 16 + lr;
      *(v8h*)(&sK[buf][r * 64 + lc]) = rk[i];
      for (int j = 0; j < 8; ++j) sVt[buf][(lc + j) * 64 + r] = rv[i][j];
    }
    __syncthreads();

    // --- issue global loads for tile kb+1 ---
    if (kb + 1 < NT) {
      const size_t baseN = ((size_t)b * L + (kb + 1) * 64) * D + h * DH;
      for (int i = 0; i < 4; ++i) {
        int r = i * 16 + lr;
        rk[i] = *(const v8h*)(Kp + baseN + (size_t)r * D + lc);
        rv[i] = *(const v8h*)(Vp + baseN + (size_t)r * D + lc);
      }
    }

    // S[16q x 64k] = Q . K^T  (contraction over Dh=64 -> 2 WMMA k-steps)
    v16h aQ[2];
    for (int cc = 0; cc < 2; ++cc) {
      int row = wave * 16 + lrow;
      int ks = cc * 32 + lhalf * 8;
      aQ[cc] = load_frag(sQ, row * 64 + ks, row * 64 + ks + 16);
    }
    v8f S[4];
    for (int t = 0; t < 4; ++t) {
      v8f cf = {};
      for (int cc = 0; cc < 2; ++cc) {
        int key = t * 16 + lrow;
        int dd = cc * 32 + lhalf * 8;
        v16h bK = load_frag(sK[buf], key * 64 + dd, key * 64 + dd + 16);
        cf = WMMA_F32_F16(aQ[cc], bK, cf);
      }
      for (int j = 0; j < 8; ++j) cf[j] += offs;
      S[t] = cf;
    }

    // online softmax update (row stats via DPP, no LDS)
    float alpha[8];
    for (int j = 0; j < 8; ++j) {
      float mv = fmaxf(fmaxf(S[0][j], S[1][j]), fmaxf(S[2][j], S[3][j]));
      mv = row_max16(mv);
      float mn = fmaxf(mprev[j], mv);
      alpha[j] = __expf(mprev[j] - mn);
      mprev[j] = mn;
    }
    float rs[8];
    for (int j = 0; j < 8; ++j) rs[j] = 0.f;
    for (int t = 0; t < 4; ++t)
      for (int j = 0; j < 8; ++j) {
        float p = __expf(S[t][j] - mprev[j]);
        S[t][j] = p;
        rs[j] += p;
      }
    for (int j = 0; j < 8; ++j)
      lsum[j] = lsum[j] * alpha[j] + row_sum16(rs[j]);

    // P (C-frag layout) -> f16 in wave-private LDS (A-frag layout readback)
    for (int t = 0; t < 4; ++t)
      for (int j = 0; j < 8; ++j) {
        int m = j + 8 * lhalf;
        sP[wbase + m * 64 + t * 16 + lrow] = (_Float16)S[t][j];
      }
    asm volatile("s_wait_dscnt 0" ::: "memory");  // intra-wave LDS RAW

    // O = O*alpha + P @ V  (contraction over 64 keys -> 2 WMMA k-steps)
    for (int t = 0; t < 4; ++t)
      for (int j = 0; j < 8; ++j) O[t][j] *= alpha[j];
    for (int cc = 0; cc < 2; ++cc) {
      int ks = cc * 32 + lhalf * 8;
      v16h aP = load_frag(sP + wbase, lrow * 64 + ks, lrow * 64 + ks + 16);
      for (int t = 0; t < 4; ++t) {
        int dcol = t * 16 + lrow;
        int ko = cc * 32 + lhalf * 8;
        v16h bV = load_frag(sVt[buf], dcol * 64 + ko, dcol * 64 + ko + 16);
        O[t] = WMMA_F32_F16(aP, bV, O[t]);
      }
    }
  }

  // normalize and store merged-head f16 output [b, q, h*64 + d]
  for (int j = 0; j < 8; ++j) lsum[j] = 1.0f / lsum[j];
  for (int t = 0; t < 4; ++t)
    for (int j = 0; j < 8; ++j) {
      int mg = qblk * 64 + wave * 16 + j + 8 * lhalf;
      int ng = h * DH + t * 16 + lrow;
      Outp[((size_t)b * L + mg) * D + ng] = (_Float16)(O[t][j] * lsum[j]);
    }
}

// ---------------------------------------------------------------------------
extern "C" void kernel_launch(void* const* d_in, const int* in_sizes, int n_in,
                              void* d_out, int out_size, void* d_ws, size_t ws_size,
                              hipStream_t stream) {
  const float* q  = (const float*)d_in[0];
  const float* k  = (const float*)d_in[1];
  const float* v  = (const float*)d_in[2];
  const float* Wq = (const float*)d_in[3];
  const float* bq = (const float*)d_in[4];
  const float* Wk = (const float*)d_in[5];
  const float* bk = (const float*)d_in[6];
  const float* Wv = (const float*)d_in[7];
  const float* bv = (const float*)d_in[8];
  const float* Wo = (const float*)d_in[9];
  const float* bo = (const float*)d_in[10];
  const float* offset = (const float*)d_in[11];

  const int M = 4 * 2048, N = 1024, K = 1024;
  const size_t SLAB = (size_t)M * N * sizeof(_Float16);   // 16.78 MB
  _Float16* Qp = (_Float16*)d_ws;
  _Float16* Kp = (_Float16*)((char*)d_ws + SLAB);
  _Float16* Vp = (_Float16*)((char*)d_ws + 2 * SLAB);
  _Float16* At = (_Float16*)((char*)d_ws + 3 * SLAB);

  dim3 ggrid(N / 64, M / 128);   // (16, 64)
  const float qscale = 0.125f;   // (QK_DIM/NUM_HEADS)^-0.5 = 64^-0.5

  gemm_bias_kernel<true, false><<<ggrid, 256, 0, stream>>>(q, Wq, bq, qscale, Qp, M, N, K);
  gemm_bias_kernel<true, false><<<ggrid, 256, 0, stream>>>(k, Wk, bk, 1.0f,   Kp, M, N, K);
  gemm_bias_kernel<true, false><<<ggrid, 256, 0, stream>>>(v, Wv, bv, 1.0f,   Vp, M, N, K);

  attn_kernel<<<4 * 16 * (2048 / 64), 128, 0, stream>>>(Qp, Kp, Vp, At, offset);

  gemm_bias_kernel<false, true><<<ggrid, 256, 0, stream>>>(At, Wo, bo, 1.0f, d_out, M, N, K);
}
